// RetinaNetPostProcessor_47674136985807
// MI455X (gfx1250) — 8-bit matrix operand order repaired and checked by reading, compiled, not run on hardware
//
#include <hip/hip_runtime.h>
#include <stdint.h>

#define NIMG    8
#define AANCH   9
#define CCLS    80
#define HH      100
#define WWID    100
#define HWA     90000
#define PER_IMG 7200000
#define ECOUNT  57600000LL
#define NBINS   2048
#define KTOP    1000
#define KPAD    1024
#define KOUT    100

#define LOGIT_THRESH (-2.9444389791664403f)
#define BBOX_CLIP    (4.135166556742356f)
#define IMG_MAX      799.0f

#define OFF_HIST   0
#define OFF_CUT    16384
#define OFF_TAKE   16392
#define OFF_TIE    16400
#define OFF_CNT    16408
#define OFF_VALID  16416
#define OFF_KEEP   16672
#define OFF_CAND   16928
#define OFF_BOXES  33312
#define OFF_SCORE  66080
#define OFF_LABEL  74272
#define OFF_SAME   82464
#define OFF_SUP    344608
#define ZERO_LO    33312

__device__ __forceinline__ uint32_t fkey(float f) {
  uint32_t b = __float_as_uint(f);
  return b ^ ((uint32_t)((int32_t)b >> 31) | 0x80000000u);
}
__device__ __forceinline__ float fkeyinv(uint32_t k) {
  uint32_t b = (k & 0x80000000u) ? (k ^ 0x80000000u) : ~k;
  return __uint_as_float(b);
}

__device__ __forceinline__ void async_load_b128_to_lds(uint32_t lds_off, const void* g) {
  asm volatile("global_load_async_to_lds_b128 %0, %1, off"
               :: "v"(lds_off), "v"((uint64_t)(uintptr_t)g) : "memory");
}
__device__ __forceinline__ void wait_async_le1() {
  asm volatile("s_wait_asynccnt 1" ::: "memory");
}
__device__ __forceinline__ void wait_async_0() {
  asm volatile("s_wait_asynccnt 0" ::: "memory");
}

__global__ void init_kernel(uint32_t* ws, float* out) {
  int i = blockIdx.x * blockDim.x + threadIdx.x;
  int stride = gridDim.x * blockDim.x;
  for (int k = i; k < ZERO_LO; k += stride) ws[k] = 0u;
  for (int k = i; k < NIMG * KPAD * 32; k += stride) ws[OFF_SAME + k] = 0u;
  for (int k = i; k < (NIMG * KOUT * 6); k += stride) out[k] = 0.0f;
}

__global__ __launch_bounds__(256) void hist_kernel(const float* __restrict__ cls,
                                                   uint32_t* __restrict__ hist) {
  __shared__ uint32_t lh[NBINS];
  __shared__ float stage[2][1024];
  const int BPI = 128;
  const int CHUNK = PER_IMG / BPI;
  int n  = blockIdx.x / BPI;
  int bi = blockIdx.x % BPI;
  int start = bi * CHUNK;
  int end   = start + CHUNK;
  int t = threadIdx.x;

  for (int b = t; b < NBINS; b += 256) lh[b] = 0u;
  __syncthreads();

  int tiles = (CHUNK + 1023) / 1024;
  const size_t imgBase = (size_t)n * PER_IMG;

  auto issue = [&](int buf, int tile) {
    int e = start + tile * 1024 + t * 4;
    if (e > PER_IMG - 4) e = PER_IMG - 4;
    uint32_t lds = (uint32_t)(uintptr_t)(&stage[buf][t * 4]);
    async_load_b128_to_lds(lds, cls + imgBase + (size_t)e);
  };

  issue(0, 0);
  for (int tl = 0; tl < tiles; ++tl) {
    if (tl + 1 < tiles) { issue((tl + 1) & 1, tl + 1); wait_async_le1(); }
    else                { wait_async_0(); }
    int base = start + tl * 1024 + t * 4;
#pragma unroll
    for (int e = 0; e < 4; ++e) {
      if (base + e < end) {
        float x = stage[tl & 1][t * 4 + e];
        if (x > LOGIT_THRESH) {
          uint32_t b = fkey(x) >> 21;
          atomicAdd(&lh[b], 1u);
        }
      }
    }
  }
  __syncthreads();
  for (int b = t; b < NBINS; b += 256) {
    uint32_t v = lh[b];
    if (v) atomicAdd(&hist[n * NBINS + b], v);
  }
}

__global__ __launch_bounds__(256) void cutoff_kernel(const uint32_t* __restrict__ hist,
                                                     uint32_t* cut, uint32_t* take) {
  __shared__ uint32_t ssum[256];
  int n = blockIdx.x, t = threadIdx.x;
  uint32_t local = 0;
#pragma unroll
  for (int k = 0; k < 8; ++k) local += hist[n * NBINS + (2047 - t * 8 - k)];
  ssum[t] = local;
  __syncthreads();
  for (int off = 1; off < 256; off <<= 1) {
    uint32_t v = ssum[t];
    if (t >= off) v += ssum[t - off];
    __syncthreads();
    ssum[t] = v;
    __syncthreads();
  }
  uint32_t inc = ssum[t], before = inc - local, total = ssum[255];
  if (total < KTOP) {
    if (t == 0) { cut[n] = 0u; take[n] = 0x7fffffffu; }
  } else if (before < KTOP && inc >= KTOP) {
    uint32_t cum = before;
#pragma unroll
    for (int k = 0; k < 8; ++k) {
      int b = 2047 - t * 8 - k;
      uint32_t h = hist[n * NBINS + b];
      if (cum + h >= KTOP) { cut[n] = (uint32_t)b; take[n] = KTOP - cum; break; }
      cum += h;
    }
  }
}

__global__ __launch_bounds__(256) void compact_kernel(const float* __restrict__ cls,
                                                      const uint32_t* __restrict__ cut,
                                                      const uint32_t* __restrict__ take,
                                                      uint32_t* tie, uint32_t* cnt,
                                                      unsigned long long* cand) {
  long long i = (long long)blockIdx.x * blockDim.x + threadIdx.x;
  long long stride = (long long)gridDim.x * blockDim.x;
  for (long long e = i * 4; e < ECOUNT; e += stride * 4) {
    __builtin_prefetch(cls + e + stride * 4, 0, 1);
    float4 v = *(const float4*)(cls + e);
    const float vv[4] = {v.x, v.y, v.z, v.w};
#pragma unroll
    for (int j = 0; j < 4; ++j) {
      float x = vv[j];
      if (x > LOGIT_THRESH) {
        long long idx = e + j;
        int n = (int)(idx / PER_IMG);
        uint32_t key = fkey(x), b = key >> 21, cb = cut[n];
        if (b >= cb) {
          bool ok = true;
          if (b == cb) ok = (atomicAdd(&tie[n], 1u) < take[n]);
          if (ok) {
            uint32_t slot = atomicAdd(&cnt[n], 1u);
            if (slot < KPAD) {
              int wi  = (int)(idx - (long long)n * PER_IMG);
              int ac  = wi / (HH * WWID);
              int pix = wi % (HH * WWID);
              int h = pix / WWID, w = pix % WWID;
              int a = ac / CCLS,  c = ac % CCLS;
              uint32_t flat = (uint32_t)(((h * WWID + w) * AANCH + a) * CCLS + c);
              cand[n * KPAD + slot] = ((unsigned long long)key << 32) | flat;
            }
          }
        }
      }
    }
  }
}

__global__ __launch_bounds__(1024) void sort_kernel(unsigned long long* cand) {
  __shared__ unsigned long long s[KPAD];
  int n = blockIdx.x, t = threadIdx.x;
  s[t] = cand[n * KPAD + t];
  __syncthreads();
  for (int k = 2; k <= KPAD; k <<= 1) {
    for (int j = k >> 1; j > 0; j >>= 1) {
      int ixj = t ^ j;
      if (ixj > t) {
        unsigned long long a = s[t], b = s[ixj];
        bool descend = ((t & k) == 0);
        if (descend ? (a < b) : (a > b)) { s[t] = b; s[ixj] = a; }
      }
      __syncthreads();
    }
  }
  cand[n * KPAD + t] = s[t];
}

__global__ __launch_bounds__(1024) void decode_kernel(const float* __restrict__ reg,
                                                      const float* __restrict__ anchors,
                                                      const unsigned long long* __restrict__ cand,
                                                      float* boxes, float* scores,
                                                      int* labels, uint32_t* validWords) {
  int n = blockIdx.x, t = threadIdx.x;
  unsigned long long p = cand[n * KPAD + t];
  uint32_t key = (uint32_t)(p >> 32), flat = (uint32_t)p;
  bool valid = key > fkey(LOGIT_THRESH);
  float bx[4] = {0.f, 0.f, 0.f, 0.f};
  float sc = 0.f; int lb = 0;
  if (valid) {
    int loc = (int)(flat / CCLS), c = (int)(flat % CCLS);
    lb = c + 1;
    int a = loc % AANCH, hw = loc / AANCH, h = hw / WWID, w = hw % WWID;
    float r[4], an[4];
#pragma unroll
    for (int j = 0; j < 4; ++j) {
      r[j]  = reg[(((size_t)n * (AANCH * 4) + a * 4 + j) * HH + h) * WWID + w];
      an[j] = anchors[((size_t)n * HWA + loc) * 4 + j];
    }
    float widths = an[2] - an[0] + 1.f, heights = an[3] - an[1] + 1.f;
    float cx = an[0] + 0.5f * widths,  cy = an[1] + 0.5f * heights;
    float dx = r[0] * 0.1f, dy = r[1] * 0.1f;
    float dw = fminf(r[2] * 0.2f, BBOX_CLIP), dh = fminf(r[3] * 0.2f, BBOX_CLIP);
    float pcx = dx * widths + cx, pcy = dy * heights + cy;
    float pw = expf(dw) * widths, ph = expf(dh) * heights;
    bx[0] = fminf(fmaxf(pcx - 0.5f * pw, 0.f), IMG_MAX);
    bx[1] = fminf(fmaxf(pcy - 0.5f * ph, 0.f), IMG_MAX);
    bx[2] = fminf(fmaxf(pcx + 0.5f * pw - 1.f, 0.f), IMG_MAX);
    bx[3] = fminf(fmaxf(pcy + 0.5f * ph - 1.f, 0.f), IMG_MAX);
    if ((bx[2] - bx[0] + 1.f) < 0.f || (bx[3] - bx[1] + 1.f) < 0.f) valid = false;
    float lg = fkeyinv(key);
    sc = 1.f / (1.f + expf(-lg));
  }
#pragma unroll
  for (int j = 0; j < 4; ++j) boxes[((size_t)n * KPAD + t) * 4 + j] = bx[j];
  scores[n * KPAD + t] = valid ? sc : 0.f;
  labels[n * KPAD + t] = valid ? lb : 0;
  uint32_t m = (uint32_t)__ballot(valid);
  if ((t & 31) == 0) validWords[n * 32 + (t >> 5)] = m;
}

typedef int v8i __attribute__((ext_vector_type(8)));

__global__ __launch_bounds__(256) void same_kernel(const int* __restrict__ labels,
                                                   uint32_t* __restrict__ sameMat) {
  const int BPI = 16;
  int n    = blockIdx.x / BPI;
  int bi   = blockIdx.x % BPI;
  int wave = threadIdx.x >> 5;
  int lane = threadIdx.x & 31;
  int half = lane >> 4;
  int waveG = bi * 8 + wave;
  for (int tt = 0; tt < 32; ++tt) {
    int tile = waveG * 32 + tt;
    int I = tile >> 6, J = tile & 63;
    int rowLab = labels[n * KPAD + I * 16 + (lane & 15)];
    int colLab = labels[n * KPAD + J * 16 + (lane & 15)];
    v8i acc = {0, 0, 0, 0, 0, 0, 0, 0};
#pragma unroll
    for (int chunk = 0; chunk < 2; ++chunk) {
      int rp = rowLab - chunk * 64;
      int cp = colLab - chunk * 64;
      v8i A, B;
#pragma unroll
      for (int v = 0; v < 8; ++v) {
        int ra = rp - ((v >> 1) * 16 + half * 8 + (v & 1) * 4);
        A[v] = (ra >= 0 && ra < 4) ? (1 << (8 * ra)) : 0;
        int rb = cp - ((v >> 1) * 16 + half * 8 + (v & 1) * 4);
        B[v] = (rb >= 0 && rb < 4) ? (1 << (8 * rb)) : 0;
      }
      acc = __builtin_amdgcn_wmma_i32_16x16x64_iu8(false, A, false, B, acc, false, false);
    }
#pragma unroll
    for (int r = 0; r < 8; ++r) {
      uint32_t m = (uint32_t)__ballot(acc[r] != 0);
      if (lane == 0) {
        uint32_t lo = m & 0xFFFFu, hi = m >> 16;
        int wj = J >> 1, sh = (J & 1) * 16;
        atomicOr(&sameMat[((size_t)n * KPAD + I * 16 + r)     * 32 + wj], lo << sh);
        atomicOr(&sameMat[((size_t)n * KPAD + I * 16 + r + 8) * 32 + wj], hi << sh);
      }
    }
  }
}

__global__ __launch_bounds__(1024) void iou_kernel(const float* __restrict__ boxes,
                                                   const uint32_t* __restrict__ sameMat,
                                                   uint32_t* __restrict__ supMat) {
  __shared__ float4 sb[KPAD];
  int n = blockIdx.x, t = threadIdx.x;
  sb[t] = ((const float4*)boxes)[(size_t)n * KPAD + t];
  __syncthreads();
  float4 bi = sb[t];
  float ai = (bi.z - bi.x + 1.f) * (bi.w - bi.y + 1.f);
  int iw = t >> 5, ib = t & 31;
  for (int wj = 0; wj < 32; ++wj) {
    uint32_t bits = 0;
    if (wj >= iw) {
#pragma unroll 8
      for (int b = 0; b < 32; ++b) {
        float4 bj = sb[wj * 32 + b];
        float aj = (bj.z - bj.x + 1.f) * (bj.w - bj.y + 1.f);
        float xx1 = fmaxf(bi.x, bj.x), yy1 = fmaxf(bi.y, bj.y);
        float xx2 = fminf(bi.z, bj.z), yy2 = fminf(bi.w, bj.w);
        float ww = fmaxf(xx2 - xx1 + 1.f, 0.f), hh = fmaxf(yy2 - yy1 + 1.f, 0.f);
        float inter = ww * hh;
        float iou = inter / (ai + aj - inter);
        bits |= (iou > 0.4f) ? (1u << b) : 0u;
      }
      if (wj == iw) bits &= (ib == 31) ? 0u : (0xFFFFFFFFu << (ib + 1));
      bits &= sameMat[((size_t)n * KPAD + t) * 32 + wj];
    }
    supMat[((size_t)n * KPAD + t) * 32 + wj] = bits;
  }
}

__global__ __launch_bounds__(32) void nms_reduce_kernel(const uint32_t* __restrict__ supMat,
                                                        const uint32_t* __restrict__ validWords,
                                                        uint32_t* __restrict__ keepWords) {
  int n = blockIdx.x, l = threadIdx.x;
  uint32_t keep = validWords[n * 32 + l];
  for (int i = 0; i < KPAD; ++i) {
    uint32_t sup = supMat[((size_t)n * KPAD + i) * 32 + l];
    uint32_t ow  = __shfl(keep, i >> 5, 32);
    if ((ow >> (i & 31)) & 1u) keep &= ~sup;
  }
  keepWords[n * 32 + l] = keep;
}

__global__ __launch_bounds__(1024) void output_kernel(const uint32_t* __restrict__ keepWords,
                                                      const float* __restrict__ boxes,
                                                      const float* __restrict__ scores,
                                                      const int* __restrict__ labels,
                                                      float* __restrict__ out) {
  __shared__ uint32_t kw[32];
  int n = blockIdx.x, t = threadIdx.x;
  if (t < 32) kw[t] = keepWords[n * 32 + t];
  __syncthreads();
  int tw = t >> 5, tb = t & 31;
  if (!((kw[tw] >> tb) & 1u)) return;
  int rank = 0;
  for (int w = 0; w < tw; ++w) rank += __popc(kw[w]);
  rank += __popc(kw[tw] & ((tb == 0) ? 0u : (0xFFFFFFFFu >> (32 - tb))));
  if (rank < KOUT) {
#pragma unroll
    for (int j = 0; j < 4; ++j)
      out[((size_t)n * KOUT + rank) * 4 + j] = boxes[((size_t)n * KPAD + t) * 4 + j];
    out[NIMG * KOUT * 4 + n * KOUT + rank] = scores[n * KPAD + t];
    out[NIMG * KOUT * 5 + n * KOUT + rank] = (float)labels[n * KPAD + t];
  }
}

extern "C" void kernel_launch(void* const* d_in, const int* in_sizes, int n_in,
                              void* d_out, int out_size, void* d_ws, size_t ws_size,
                              hipStream_t stream) {
  (void)in_sizes; (void)n_in; (void)out_size; (void)ws_size;
  const float* cls = (const float*)d_in[0];
  const float* reg = (const float*)d_in[1];
  const float* anc = (const float*)d_in[2];
  float* out = (float*)d_out;
  uint32_t* ws = (uint32_t*)d_ws;

  init_kernel<<<512, 256, 0, stream>>>(ws, out);
  hist_kernel<<<NIMG * 128, 256, 0, stream>>>(cls, ws + OFF_HIST);
  cutoff_kernel<<<NIMG, 256, 0, stream>>>(ws + OFF_HIST, ws + OFF_CUT, ws + OFF_TAKE);
  compact_kernel<<<2048, 256, 0, stream>>>(cls, ws + OFF_CUT, ws + OFF_TAKE,
                                           ws + OFF_TIE, ws + OFF_CNT,
                                           (unsigned long long*)(ws + OFF_CAND));
  sort_kernel<<<NIMG, KPAD, 0, stream>>>((unsigned long long*)(ws + OFF_CAND));
  decode_kernel<<<NIMG, KPAD, 0, stream>>>(reg, anc,
                                           (const unsigned long long*)(ws + OFF_CAND),
                                           (float*)(ws + OFF_BOXES), (float*)(ws + OFF_SCORE),
                                           (int*)(ws + OFF_LABEL), ws + OFF_VALID);
  same_kernel<<<NIMG * 16, 256, 0, stream>>>((const int*)(ws + OFF_LABEL), ws + OFF_SAME);
  iou_kernel<<<NIMG, KPAD, 0, stream>>>((const float*)(ws + OFF_BOXES), ws + OFF_SAME,
                                        ws + OFF_SUP);
  nms_reduce_kernel<<<NIMG, 32, 0, stream>>>(ws + OFF_SUP, ws + OFF_VALID, ws + OFF_KEEP);
  output_kernel<<<NIMG, KPAD, 0, stream>>>(ws + OFF_KEEP, (const float*)(ws + OFF_BOXES),
                                           (const float*)(ws + OFF_SCORE),
                                           (const int*)(ws + OFF_LABEL), out);
}
